// BahdanauAttention_26946624815560
// MI455X (gfx1250) — matrix operand order repaired and checked by reading, not compiled, not run on hardware
//
#include <hip/hip_runtime.h>
#include <stdint.h>

// CDNA5 WMMA / TDM vector types (wave32)
typedef __attribute__((ext_vector_type(16))) __bf16       v16bf;
typedef __attribute__((ext_vector_type(8)))  float        v8f;
typedef __attribute__((ext_vector_type(4)))  unsigned int v4u;
typedef __attribute__((ext_vector_type(4)))  int          v4i;
typedef __attribute__((ext_vector_type(8)))  int          v8i;

#define B_    32
#define T_    2048
#define D_    512
#define U_    512
#define KSTEP 32
#define MROWS 64           // T-rows per block (4 WMMA M-tiles)
#define NK    (D_ / KSTEP) // 16 K-steps

// round-to-nearest-even f32 -> bf16, packed pair {lo, hi}
static __device__ __forceinline__ uint32_t pack_bf16x2(float lo, float hi) {
    union { float f; uint32_t u; } a, b;
    a.f = lo; b.f = hi;
    uint32_t ra = a.u + 0x7FFFu + ((a.u >> 16) & 1u);
    uint32_t rb = b.u + 0x7FFFu + ((b.u >> 16) & 1u);
    return (ra >> 16) | (rb & 0xFFFF0000u);
}

// ---- Tensor Data Mover: async 2D f32 tile load Global -> LDS ----
// Builds a D# (groups 0/1 per CDNA5 ISA 8.3/8.4; groups 2/3 zero => 2D tensor)
// and issues TENSOR_LOAD_TO_LDS. Tracked by TENSORcnt.
static __device__ __forceinline__ void tdm_load_2d_f32(
    const float* gaddr, uint32_t lds_byte_addr,
    uint32_t tile_d0, uint32_t tile_d1,
    uint32_t tensor_d0, uint32_t tensor_d1,
    uint64_t stride0 /* elements */) {
    uint64_t ga = (uint64_t)(uintptr_t)gaddr;
    v4u g0;
    g0.x = 1u;                                   // count=1 (valid), user mode
    g0.y = lds_byte_addr;                        // lds_addr
    g0.z = (uint32_t)(ga & 0xFFFFFFFFu);         // global_addr[31:0]
    g0.w = (uint32_t)((ga >> 32) & 0x1FFFFFFu)   // global_addr[56:32]
         | (2u << 30);                           // type=2 ("image")
    v8i g1;
    g1[0] = (int)(2u << 16);                                   // data_size=2 -> 4B
    g1[1] = (int)((tensor_d0 & 0xFFFFu) << 16);                // tensor_dim0[15:0]
    g1[2] = (int)((tensor_d0 >> 16) | ((tensor_d1 & 0xFFFFu) << 16));
    g1[3] = (int)((tensor_d1 >> 16) | (tile_d0 << 16));        // tile_dim0
    g1[4] = (int)(tile_d1 & 0xFFFFu);                          // tile_dim1 (tile_dim2=0)
    g1[5] = (int)(stride0 & 0xFFFFFFFFu);                      // tensor_dim0_stride lo
    g1[6] = (int)((stride0 >> 32) & 0xFFFFu);                  // stride hi (stride1=0)
    g1[7] = 0;
    v4i g2 = {0, 0, 0, 0};
    v4i g3 = {0, 0, 0, 0};
#if __clang_major__ >= 23
    v8i g4 = {0, 0, 0, 0, 0, 0, 0, 0};
    __builtin_amdgcn_tensor_load_to_lds(g0, g1, g2, g3, g4, 0);
#else
    __builtin_amdgcn_tensor_load_to_lds(g0, g1, g2, g3, 0);
#endif
}

// -------- Kernel 1: pq[b,u] = (query @ W2)[b,u] + b1[u] + b2[u] --------
__global__ __launch_bounds__(256)
void proj_q_kernel(const float* __restrict__ query,
                   const float* __restrict__ W2,
                   const float* __restrict__ b1,
                   const float* __restrict__ b2,
                   float* __restrict__ pq) {
    const int b = blockIdx.x;
    const float* q = query + b * D_;
    for (int u = threadIdx.x; u < U_; u += blockDim.x) {
        float acc = b1[u] + b2[u];
        for (int d = 0; d < D_; ++d)
            acc += q[d] * W2[d * U_ + u];
        pq[b * U_ + u] = acc;
    }
}

// -------- Kernel 2: scores[b,t] = tanh(values@W1 + pq) . V + bV --------
// Block = 256 threads = 8 waves; handles (b, 64 T-rows).
// TDM streams raw f32 slabs of W1 (32x512) and values (64x32) into LDS,
// overlapped with WMMA on the previous (bf16-packed) slab.
// Wave w covers U columns [w*64, w*64+64) as 4 N-tiles x 4 M-tiles.
__global__ __launch_bounds__(256)
void score_kernel(const float* __restrict__ values,
                  const float* __restrict__ W1,
                  const float* __restrict__ V,
                  const float* __restrict__ bV,
                  const float* __restrict__ pq,
                  float* __restrict__ scores) {
    __shared__ float    rawW[KSTEP * U_];            // 64 KB  (TDM dest)
    __shared__ float    rawV[MROWS * KSTEP];         //  8 KB  (TDM dest)
    __shared__ uint32_t w1T[(KSTEP / 2) * U_];       // 32 KB  packed bf16 pairs
    __shared__ uint32_t valsT[MROWS * (KSTEP / 2)];  //  4 KB  packed bf16 pairs
    __shared__ float    scoreAcc[MROWS];

    const int tid  = threadIdx.x;
    const int lane = tid & 31;
    const int wave = tid >> 5;
    const int b    = blockIdx.y;
    const int t0   = blockIdx.x * MROWS;
    const int u0   = wave * 64;

    if (tid < MROWS) scoreAcc[tid] = 0.0f;

    const float* vbase = values + (size_t)b * T_ * D_ + (size_t)t0 * D_;

    v8f acc[4][4];  // [m-tile][n-tile]
#pragma unroll
    for (int mt = 0; mt < 4; ++mt)
#pragma unroll
        for (int n = 0; n < 4; ++n)
#pragma unroll
            for (int e = 0; e < 8; ++e) acc[mt][n][e] = 0.0f;

    const uint32_t ldsW = (uint32_t)(uintptr_t)(void*)rawW;
    const uint32_t ldsV = (uint32_t)(uintptr_t)(void*)rawV;

    // Prologue: kick off TDM for slab 0 (wave 0 issues; EXEC-independent).
    if (wave == 0) {
        tdm_load_2d_f32(W1, ldsW, U_, KSTEP, U_, KSTEP, U_);
        tdm_load_2d_f32(vbase, ldsV, KSTEP, MROWS, D_, MROWS, D_);
    }

    for (int ks = 0; ks < NK; ++ks) {
        const int k0 = ks * KSTEP;
        __builtin_amdgcn_s_wait_tensorcnt(0);   // wave0: raw slab ks landed
        __syncthreads();

        // Convert raw f32 slabs -> packed bf16-pair tiles.
        // W1: w1T[kp*U_+u] = {W1[k0+2kp, u], W1[k0+2kp+1, u]}
        for (int i = 0; i < 32; ++i) {
            int idx = i * 256 + tid;       // 0..8191
            int kp  = idx >> 9;
            int u   = idx & (U_ - 1);
            w1T[kp * U_ + u] =
                pack_bf16x2(rawW[(2 * kp) * U_ + u], rawW[(2 * kp + 1) * U_ + u]);
        }
        // values: valsT[row*16+kp] = {vals[row, k0+2kp], vals[row, k0+2kp+1]}
#pragma unroll
        for (int i = 0; i < 4; ++i) {
            int idx = i * 256 + tid;       // 0..1023
            int row = idx >> 4;
            int kp  = idx & 15;
            valsT[row * 16 + kp] =
                pack_bf16x2(rawV[row * KSTEP + 2 * kp], rawV[row * KSTEP + 2 * kp + 1]);
        }
        __syncthreads();                   // packed ready; raw buffers free

        // Overlap: stream next raw slab while we run WMMA on this one.
        if (wave == 0 && ks + 1 < NK) {
            const int kn = k0 + KSTEP;
            tdm_load_2d_f32(W1 + (size_t)kn * U_, ldsW, U_, KSTEP, U_, KSTEP, U_);
            tdm_load_2d_f32(vbase + kn, ldsV, KSTEP, MROWS, D_, MROWS, D_);
        }

        // A fragments (16x32 bf16, one per M-tile): lanes 0-15 rows M=0..15
        // hold K-pairs 0..3 (v0-3) and 8..11 (v4-7); lanes 16-31 get +4.
        union { v16bf v; uint32_t u[8]; } afrag[4];
        {
            int m  = lane & 15;
            int hk = (lane >> 4) << 2;
#pragma unroll
            for (int mt = 0; mt < 4; ++mt) {
                int row = mt * 16 + m;
#pragma unroll
                for (int i = 0; i < 8; ++i) {
                    int kp = (i & 3) + ((i >> 2) << 3) + hk;
                    afrag[mt].u[i] = valsT[row * 16 + kp];
                }
            }
        }

        // B fragment per N-tile (32x16 bf16): VGPR j holds K-pair
        // kp = (j&3) + 8*(j>>2) (+4 for lanes 16-31), column N = lane&15.
#pragma unroll
        for (int n = 0; n < 4; ++n) {
            union { v16bf v; uint32_t u[8]; } bfrag;
            int col   = u0 + n * 16 + (lane & 15);
            int kbase = (lane >> 4) << 2;
#pragma unroll
            for (int j = 0; j < 8; ++j)
                bfrag.u[j] = w1T[((j & 3) + ((j >> 2) << 3) + kbase) * U_ + col];
#pragma unroll
            for (int mt = 0; mt < 4; ++mt)
                acc[mt][n] = __builtin_amdgcn_wmma_f32_16x16x32_bf16(
                    false, afrag[mt].v, false, bfrag.v, (short)0, acc[mt][n],
                    false, false);
        }
    }

    // Epilogue: s[row] = sum_u tanh(proj_v[row,u] + pq[b,u]) * V[u]
    // C layout: lane l, VGPR r -> (M = r + (l>=16 ? 8:0), N = l&15).
#pragma unroll
    for (int n = 0; n < 4; ++n) {
        int u    = u0 + n * 16 + (lane & 15);
        float p  = pq[b * U_ + u];
        float vu = V[u];
        int mhi  = (lane >> 4) << 3;
#pragma unroll
        for (int mt = 0; mt < 4; ++mt) {
#pragma unroll
            for (int r = 0; r < 8; ++r) {
                float s = tanhf(acc[mt][n][r] + p) * vu;
#pragma unroll
                for (int off = 1; off < 16; off <<= 1)
                    s += __shfl_xor(s, off, 32);
                if ((lane & 15) == 0)
                    atomicAdd(&scoreAcc[mt * 16 + r + mhi], s);  // ds_add_f32
            }
        }
    }
    __syncthreads();
    if (tid < MROWS)
        scores[b * T_ + t0 + tid] = scoreAcc[tid] + bV[0];
}

// -------- Kernel 3: softmax over T per batch --------
__global__ __launch_bounds__(256)
void softmax_kernel(const float* __restrict__ scores, float* __restrict__ attn) {
    __shared__ float red[256];
    const int b = blockIdx.x, tid = threadIdx.x;
    const float* s = scores + b * T_;

    float m = -3.4e38f;
    for (int t = tid; t < T_; t += 256) m = fmaxf(m, s[t]);
    red[tid] = m; __syncthreads();
    for (int o = 128; o > 0; o >>= 1) {
        if (tid < o) red[tid] = fmaxf(red[tid], red[tid + o]);
        __syncthreads();
    }
    m = red[0]; __syncthreads();

    float sum = 0.0f;
    for (int t = tid; t < T_; t += 256) sum += __expf(s[t] - m);
    red[tid] = sum; __syncthreads();
    for (int o = 128; o > 0; o >>= 1) {
        if (tid < o) red[tid] += red[tid + o];
        __syncthreads();
    }
    float inv = 1.0f / red[0];

    for (int t = tid; t < T_; t += 256)
        attn[b * T_ + t] = __expf(s[t] - m) * inv;
}

// -------- Kernel 4: context[b,d] = sum_t attn[b,t] * values[b,t,d] --------
__global__ __launch_bounds__(256)
void context_kernel(const float* __restrict__ values,
                    const float* __restrict__ attn,
                    float* __restrict__ ctx) {
    const int b = blockIdx.y;
    const int d = blockIdx.x * 256 + threadIdx.x;
    const float* vb = values + (size_t)b * T_ * D_ + d;
    const float* ab = attn + b * T_;
    float acc = 0.0f;
    for (int t = 0; t < T_; ++t) {
        __builtin_prefetch(vb + (size_t)(t + 8) * D_, 0, 1);  // global_prefetch_b8
        acc += ab[t] * vb[(size_t)t * D_];
    }
    ctx[b * D_ + d] = acc;
}

extern "C" void kernel_launch(void* const* d_in, const int* in_sizes, int n_in,
                              void* d_out, int out_size, void* d_ws, size_t ws_size,
                              hipStream_t stream) {
    const float* values = (const float*)d_in[0];  // [B,T,D]
    const float* query  = (const float*)d_in[1];  // [B,D]
    const float* W1     = (const float*)d_in[2];  // [D,U]
    const float* b1     = (const float*)d_in[3];  // [U]
    const float* W2     = (const float*)d_in[4];  // [D,U]
    const float* b2     = (const float*)d_in[5];  // [U]
    const float* V      = (const float*)d_in[6];  // [U,1]
    const float* bV     = (const float*)d_in[7];  // [1]
    (void)in_sizes; (void)n_in; (void)out_size; (void)ws_size;

    float* ctx  = (float*)d_out;                  // [B,D]
    float* attn = (float*)d_out + B_ * D_;        // [B,T,1]
    float* pq     = (float*)d_ws;                 // [B,U]
    float* scores = (float*)d_ws + B_ * U_;       // [B,T]

    proj_q_kernel<<<dim3(B_), dim3(256), 0, stream>>>(query, W2, b1, b2, pq);
    score_kernel<<<dim3(T_ / MROWS, B_), dim3(256), 0, stream>>>(values, W1, V, bV, pq, scores);
    softmax_kernel<<<dim3(B_), dim3(256), 0, stream>>>(scores, attn);
    context_kernel<<<dim3(D_ / 256, B_), dim3(256), 0, stream>>>(values, attn, ctx);
}
